// MPNEncoder_82858509074740
// MI455X (gfx1250) — matrix operand order repaired and checked by reading, not compiled, not run on hardware
//
#include <hip/hip_runtime.h>
#include <hip/hip_bf16.h>

typedef __attribute__((ext_vector_type(16))) __bf16 v16bf;
typedef __attribute__((ext_vector_type(8)))  float  v8f;

__device__ __forceinline__ unsigned short f2bf(float f) {
    unsigned int u = __builtin_bit_cast(unsigned int, f);
    u += 0x7FFFu + ((u >> 16) & 1u);          // round-to-nearest-even
    return (unsigned short)(u >> 16);
}

union Frag16 { uint4 q[2]; v16bf v; };
union Pack8  { unsigned short s[8]; uint4 q; };
union Pack4  { unsigned short s[4]; uint2 q; };

// ---------------------------------------------------------------------------
// GEMM: Y[M,300] = A(bf16,[Mtiles*16,Kp]) @ W (bf16, stored transposed as
// WT[320][Kp], zero-padded).  Each wave: one 16x64 strip (4 n-tiles).
//   if Yraw  : Yraw  = X
//   if Yrelu : Yrelu = relu(X + (Aadd?Aadd:0) + (bias?bias:0))
// ---------------------------------------------------------------------------
#define NSTRIP 4
__global__ __launch_bounds__(128)
void gemm_wmma_bf16(const unsigned short* __restrict__ A,
                    const unsigned short* __restrict__ WT,
                    int Mtiles, int Kp, int N,
                    float* __restrict__ Yraw,
                    const float* __restrict__ Aadd,
                    const float* __restrict__ bias,
                    float* __restrict__ Yrelu)
{
    const int wave  = threadIdx.x >> 5;
    const int lane  = threadIdx.x & 31;
    const int mtile = blockIdx.x * 4 + wave;
    if (mtile >= Mtiles) return;               // wave-uniform, EXEC stays full

    const int half   = lane >> 4;              // 0: lanes 0-15, 1: lanes 16-31
    const int l16    = lane & 15;
    const int n0tile = blockIdx.y * NSTRIP;

    // A fragment base: lane<16 -> K runs {kb, kb+16}; lane>=16 -> {kb+8, kb+24}
    const unsigned short* Ap = A + (size_t)(mtile * 16 + l16) * Kp + half * 8;

    // B fragment base per n-tile: WT row = output column; lane<16 -> K runs {kb, kb+16}; lane>=16 -> {kb+8, kb+24}
    const unsigned short* Bp[NSTRIP];
#pragma unroll
    for (int j = 0; j < NSTRIP; ++j)
        Bp[j] = WT + (size_t)((n0tile + j) * 16 + l16) * Kp + half * 8;

    v8f acc[NSTRIP];
#pragma unroll
    for (int j = 0; j < NSTRIP; ++j) acc[j] = (v8f){};

    for (int kb = 0; kb < Kp; kb += 32) {
        if (kb + 32 < Kp) __builtin_prefetch(Ap + kb + 32, 0, 3);

        Frag16 af;
        af.q[0] = *(const uint4*)(Ap + kb);
        af.q[1] = *(const uint4*)(Ap + kb + 16);

#pragma unroll
        for (int j = 0; j < NSTRIP; ++j) {
            Frag16 bf;
            bf.q[0] = *(const uint4*)(Bp[j] + kb);
            bf.q[1] = *(const uint4*)(Bp[j] + kb + 16);
            acc[j] = __builtin_amdgcn_wmma_f32_16x16x32_bf16(
                         false, af.v, false, bf.v, (short)0, acc[j], false, false);
        }
    }

    // D layout: VGPR r, lanes<16 -> M=r, lanes>=16 -> M=r+8; col = l16
#pragma unroll
    for (int j = 0; j < NSTRIP; ++j) {
        const int ocol = (n0tile + j) * 16 + l16;
        if (ocol >= N) continue;
#pragma unroll
        for (int r = 0; r < 8; ++r) {
            const int orow = mtile * 16 + half * 8 + r;
            const size_t idx = (size_t)orow * N + ocol;
            float x = acc[j][r];
            if (Yraw)  Yraw[idx] = x;
            if (Aadd)  x += Aadd[idx];
            if (bias)  x += bias[ocol];
            if (Yrelu) Yrelu[idx] = x > 0.0f ? x : 0.0f;
        }
    }
}

// ---------------------------------------------------------------------------
// Pre-processing / conversion kernels
// ---------------------------------------------------------------------------

// W f32[K,N] -> WT bf16[320][Kp] (transposed, zero-padded)
__global__ void cvt_WT(const float* __restrict__ W, unsigned short* __restrict__ WT,
                       int K, int N, int Kp)
{
    const int i = blockIdx.x * blockDim.x + threadIdx.x;
    if (i >= 320 * Kp) return;
    const int n = i / Kp, k = i % Kp;
    WT[i] = (k < K && n < N) ? f2bf(W[(size_t)k * N + n]) : (unsigned short)0;
}

// A f32[M,K] -> bf16[M,Kp] zero-padded (8 outputs / thread, b128 stores)
__global__ void cvt_pad_A(const float* __restrict__ in, unsigned short* __restrict__ out,
                          int M, int K, int Kp)
{
    const int per = Kp >> 3;
    const int i = blockIdx.x * blockDim.x + threadIdx.x;
    if (i >= M * per) return;
    const int r = i / per, j8 = (i % per) * 8;
    Pack8 p;
#pragma unroll
    for (int e = 0; e < 8; ++e) {
        const int k = j8 + e;
        p.s[e] = (k < K) ? f2bf(in[(size_t)r * K + k]) : (unsigned short)0;
    }
    *(uint4*)(out + (size_t)r * Kp + j8) = p.q;
}

// out[a,:] = sum_{j<6} msg[a2b[a,j], :]   (f32, float4 vectorized)
__global__ void aggregate6(const float* __restrict__ msg, const int* __restrict__ a2b,
                           float* __restrict__ out, int n_atoms, int H)
{
    const int H4 = H >> 2;
    const int i = blockIdx.x * blockDim.x + threadIdx.x;
    if (i >= n_atoms * H4) return;
    const int a = i / H4, h4 = (i % H4) * 4;
    const int* nb = a2b + (size_t)a * 6;
    float4 s = make_float4(0.f, 0.f, 0.f, 0.f);
#pragma unroll
    for (int j = 0; j < 6; ++j) {
        const float4 v = *(const float4*)(msg + (size_t)nb[j] * H + h4);
        s.x += v.x; s.y += v.y; s.z += v.z; s.w += v.w;
    }
    *(float4*)(out + (size_t)a * H + h4) = s;
}

// Ab[b,k] = bf16( amsg[b2a[b],k] - msg[b2revb[b],k] ), k<H; zero-pad to Kp
__global__ void bond_prep_bf16(const float* __restrict__ amsg, const float* __restrict__ msg,
                               const int* __restrict__ b2a, const int* __restrict__ b2revb,
                               unsigned short* __restrict__ out, int n_bonds, int H, int Kp)
{
    const int per = Kp >> 2;
    const int i = blockIdx.x * blockDim.x + threadIdx.x;
    if (i >= n_bonds * per) return;
    const int b = i / per, j4 = (i % per) * 4;
    Pack4 p;
    if (j4 < H) {
        const float4 u = *(const float4*)(amsg + (size_t)b2a[b]    * H + j4);
        const float4 v = *(const float4*)(msg  + (size_t)b2revb[b] * H + j4);
        p.s[0] = f2bf(u.x - v.x); p.s[1] = f2bf(u.y - v.y);
        p.s[2] = f2bf(u.z - v.z); p.s[3] = f2bf(u.w - v.w);
    } else {
        p.s[0] = p.s[1] = p.s[2] = p.s[3] = 0;
    }
    *(uint2*)(out + (size_t)b * Kp + j4) = p.q;
}

// Ab[a,:] = bf16( concat(f_atoms[a,:AF], amsg[a,:H]) ), zero-pad to Kp
__global__ void concat_bf16(const float* __restrict__ f_atoms, const float* __restrict__ amsg,
                            unsigned short* __restrict__ out, int n_atoms, int AF, int H, int Kp)
{
    const int i = blockIdx.x * blockDim.x + threadIdx.x;
    if (i >= n_atoms * Kp) return;
    const int a = i / Kp, c = i % Kp;
    float v = 0.0f;
    if (c < AF)          v = f_atoms[(size_t)a * AF + c];
    else if (c < AF + H) v = amsg[(size_t)a * H + (c - AF)];
    out[i] = f2bf(v);
}

// ---------------------------------------------------------------------------
extern "C" void kernel_launch(void* const* d_in, const int* in_sizes, int n_in,
                              void* d_out, int out_size, void* d_ws, size_t ws_size,
                              hipStream_t stream) {
    const float* f_atoms = (const float*)d_in[0];
    const float* f_bonds = (const float*)d_in[1];
    const int*   a2b     = (const int*)d_in[2];
    const int*   b2a     = (const int*)d_in[3];
    const int*   b2revb  = (const int*)d_in[4];
    const float* W_i     = (const float*)d_in[5];
    const float* W_h     = (const float*)d_in[6];
    const float* W_o     = (const float*)d_in[7];
    const float* b_o     = (const float*)d_in[8];

    const int AF = 133, BF = 14, H = 300;
    const int IN = AF + BF;                     // 147
    const int n_atoms = in_sizes[0] / AF;       // 100000
    const int n_bonds = in_sizes[3];            // 200000
    const int KpI = 160, KpH = 320, KpO = 448;  // K padded to 32

    char* ws = (char*)d_ws;
    const size_t SB = (size_t)n_bonds * H * sizeof(float);       // 240 MB
    const size_t SA = (size_t)n_atoms * H * sizeof(float);       // 120 MB
    const size_t AB = (size_t)n_bonds * KpH * sizeof(unsigned short); // 128 MB, max A
    float*          inp     = (float*)(ws);
    float*          message = (float*)(ws + SB);
    float*          amsg    = (float*)(ws + 2 * SB);
    unsigned short* Ab      = (unsigned short*)(ws + 2 * SB + SA);
    unsigned short* WTi     = (unsigned short*)(ws + 2 * SB + SA + AB);
    unsigned short* WTh     = WTi + (size_t)320 * KpI;
    unsigned short* WTo     = WTh + (size_t)320 * KpH;

    const dim3 blk(128);
    const dim3 gB((n_bonds / 16 + 3) / 4, 5);   // bond GEMMs: 12500 m-tiles
    const dim3 gA((n_atoms / 16 + 3) / 4, 5);   // atom GEMM:  6250 m-tiles

    // ---- weight conversion (tiny, once per launch) ----
    int t;
    t = 320 * KpI; cvt_WT<<<(t + 255) / 256, 256, 0, stream>>>(W_i, WTi, IN,      H, KpI);
    t = 320 * KpH; cvt_WT<<<(t + 255) / 256, 256, 0, stream>>>(W_h, WTh, H,       H, KpH);
    t = 320 * KpO; cvt_WT<<<(t + 255) / 256, 256, 0, stream>>>(W_o, WTo, AF + H,  H, KpO);

    // ---- 1) inp = f_bonds @ W_i ; message = relu(inp) ----
    t = n_bonds * (KpI >> 3);
    cvt_pad_A<<<(t + 255) / 256, 256, 0, stream>>>(f_bonds, Ab, n_bonds, IN, KpI);
    gemm_wmma_bf16<<<gB, blk, 0, stream>>>(Ab, WTi, n_bonds / 16, KpI, H,
                                           inp, nullptr, nullptr, message);

    // ---- 2) two message-passing iterations ----
    for (int d = 0; d < 2; ++d) {
        t = n_atoms * (H / 4);
        aggregate6<<<(t + 255) / 256, 256, 0, stream>>>(message, a2b, amsg, n_atoms, H);

        t = n_bonds * (KpH >> 2);
        bond_prep_bf16<<<(t + 255) / 256, 256, 0, stream>>>(amsg, message, b2a, b2revb,
                                                            Ab, n_bonds, H, KpH);

        // message = relu(inp + Ab @ W_h)
        gemm_wmma_bf16<<<gB, blk, 0, stream>>>(Ab, WTh, n_bonds / 16, KpH, H,
                                               nullptr, inp, nullptr, message);
    }

    // ---- 3) readout ----
    t = n_atoms * (H / 4);
    aggregate6<<<(t + 255) / 256, 256, 0, stream>>>(message, a2b, amsg, n_atoms, H);

    t = n_atoms * KpO;
    concat_bf16<<<(t + 255) / 256, 256, 0, stream>>>(f_atoms, amsg, Ab,
                                                     n_atoms, AF, H, KpO);

    gemm_wmma_bf16<<<gA, blk, 0, stream>>>(Ab, WTo, n_atoms / 16, KpO, H,
                                           nullptr, nullptr, b_o, (float*)d_out);
}
